// MGULayer_57887569215829
// MI455X (gfx1250) — matrix operand order repaired and checked by reading, not compiled, not run on hardware
//
#include <hip/hip_runtime.h>
#include <hip/hip_bf16.h>

// ---------------------------------------------------------------------------
// MGU layer for MI455X (gfx1250, wave32, WMMA), persistent-scan version.
//   B=64, T=512, DIN=512, DOUT=1024
// Per step (M=64, K=1536, N=1024 each):
//   f  = sigmoid([h, x_t] @ W_f + b_f)
//   h~ = tanh   ([f*h, x_t] @ W_h + b_h)
//   h  = (1-f)*h + f*h~
// One persistent kernel runs all 512 steps; 64 WGs sync via a device-scope
// split (arrive/wait) generation barrier, 2/step. The x-projection part of
// each gate GEMM is independent of the barrier, so it executes between
// arrive and wait, hiding the barrier's atomic round-trip latency.
// h and f live in VGPRs for the whole scan. Each WG owns one 16-column
// N-tile and caches BOTH gates' W^T rows in 96 KB LDS (one-time 6 MB fill):
// every B-fragment of every step is a ds_load_b128.
// bf16 inputs / f32 accumulation via v_wmma_f32_16x16x32_bf16.
// ---------------------------------------------------------------------------

typedef __bf16 bf16_t;
typedef __attribute__((ext_vector_type(16))) __bf16 v16bf;
typedef __attribute__((ext_vector_type(8)))  __bf16 v8bf;
typedef __attribute__((ext_vector_type(8)))  float   v8f;

#define NB    64      // batch
#define NT    512     // time steps
#define NDIN  512
#define NDOUT 1024
#define NKTOT 1536    // DOUT (h part) + DIN (x part)
#define NWG   64      // persistent workgroups (one per N-tile)
#define NK8   (NKTOT / 8)

union BF16x16 { v16bf v; v8bf h[2]; };

// A fragment (16x32 bf16, MxK): lane holds row mb+(l&15);
// elems 0..7  = K [kc + hi*8,      +8) ; elems 8..15 = K [kc + hi*8 + 16, +8)
__device__ __forceinline__ v16bf load_frag_a(const bf16_t* rowp, int kc, int lane) {
  const int hi = lane >> 4;
  const bf16_t* p = rowp + kc + hi * 8;
  BF16x16 f;
  f.h[0] = *(const v8bf*)(p);
  f.h[1] = *(const v8bf*)(p + 16);
  return f.v;
}

// B fragment (32x16 bf16, KxN) from a K-contiguous row of this lane's column
// (rowp = row base for col l&15): elems 0..7 = K [kc + hi*8, +8) ; elems 8..15 = K [kc + hi*8 + 16, +8).
// rowp points into LDS; addrspace inference lowers these to ds_load_b128.
__device__ __forceinline__ v16bf load_frag_b(const bf16_t* rowp, int kc, int lane) {
  const int hi = lane >> 4;
  const bf16_t* p = rowp + kc + hi * 8;
  BF16x16 f;
  f.h[0] = *(const v8bf*)(p);
  f.h[1] = *(const v8bf*)(p + 16);
  return f.v;
}

__device__ __forceinline__ v8f wmma_bf16(v16bf a, v16bf b, v8f c) {
  return __builtin_amdgcn_wmma_f32_16x16x32_bf16(
      /*neg_a=*/false, a, /*neg_b=*/false, b,
      /*c_mod=*/(short)0, c, /*reuse_a=*/false, /*reuse_b=*/false);
}

// x-part of a gate tile: x_t(16x512) @ W_x^T cols, from-zero accumulator.
// Independent of the grid barrier (x and W are constants within the scan).
__device__ __forceinline__ v8f gemm_x(const bf16_t* __restrict__ xrow,
                                      const bf16_t* browx,   // LDS, x-part rows
                                      int lane) {
  v8f acc = {};
  #pragma unroll 4
  for (int kc = 0; kc < NDIN; kc += 32)
    acc = wmma_bf16(load_frag_a(xrow, kc, lane), load_frag_b(browx, kc, lane), acc);
  return acc;
}

// Recurrent part of a gate tile: h/g (16x1024) @ W_h^T cols, accumulate into acc.
__device__ __forceinline__ v8f gemm_h(const bf16_t* __restrict__ arow,
                                      const bf16_t* brow,    // LDS, K-contig
                                      int lane, v8f acc) {
  #pragma unroll 4
  for (int kc = 0; kc < NDOUT; kc += 32)
    acc = wmma_bf16(load_frag_a(arow, kc, lane), load_frag_b(brow, kc, lane), acc);
  return acc;
}

// --- Split device-wide generation barrier across NWG persistent WGs. -------
// arrive(): publish stores, signal; independent work may follow.
// wait():   spin until generation advances, then acquire.
__device__ __forceinline__ unsigned bar_arrive(unsigned* __restrict__ cnt,
                                               unsigned* __restrict__ gen) {
  __threadfence();                 // publish this WG's stores (device scope)
  __syncthreads();                 // all waves' stores are fenced before signal
  unsigned g = 0;
  if (threadIdx.x == 0) {
    g = __hip_atomic_load(gen, __ATOMIC_RELAXED, __HIP_MEMORY_SCOPE_AGENT);
    unsigned a = __hip_atomic_fetch_add(cnt, 1u, __ATOMIC_ACQ_REL, __HIP_MEMORY_SCOPE_AGENT);
    if (a == NWG - 1u) {
      __hip_atomic_store(cnt, 0u, __ATOMIC_RELAXED, __HIP_MEMORY_SCOPE_AGENT);
      __hip_atomic_fetch_add(gen, 1u, __ATOMIC_RELEASE, __HIP_MEMORY_SCOPE_AGENT);
    }
  }
  return g;                        // meaningful in thread 0 only
}

__device__ __forceinline__ void bar_wait(unsigned* __restrict__ gen, unsigned g) {
  if (threadIdx.x == 0) {
    while (__hip_atomic_load(gen, __ATOMIC_ACQUIRE, __HIP_MEMORY_SCOPE_AGENT) == g)
      __builtin_amdgcn_s_sleep(2);
  }
  __syncthreads();
  __threadfence();                 // acquire: invalidate L0 before reading peers' data
}

// ---------------------------------------------------------------------------
// Persistent scan kernel: grid (64), block 128 (4 waves).
// WG -> one N-tile (16 cols, cached W^T rows in LDS); wave -> one M-tile.
// Same (m,n) tile owned in both phases and all t, so h[8]/f[8] stay in VGPRs.
// ---------------------------------------------------------------------------
__global__ __launch_bounds__(128) void mgu_scan_kernel(
    const bf16_t* __restrict__ xt,   // [NB][NT][NDIN] bf16
    const bf16_t* __restrict__ WfT,  // [NDOUT][NKTOT] bf16 (K-contig)
    const bf16_t* __restrict__ WhT,  // [NDOUT][NKTOT] bf16
    const float*  __restrict__ bfv_, // [NDOUT]
    const float*  __restrict__ bhv_, // [NDOUT]
    bf16_t*       __restrict__ hB,   // [NB][NDOUT] bf16 h (GEMM A input)
    bf16_t*       __restrict__ gB,   // [NB][NDOUT] bf16 f*h (GEMM A input)
    float*        __restrict__ out,  // [NB][NDOUT][NT]
    unsigned*     __restrict__ bar)  // bar[0]=count, bar[1]=generation
{
  __shared__ __align__(16) bf16_t ldsW[2][16][NKTOT];   // 96 KB: Wf + Wh cols

  const int lane = threadIdx.x & 31;
  const int wave = threadIdx.x >> 5;
  const int l15  = lane & 15;
  const int nb   = blockIdx.x * 16;      // this WG's 16 output columns
  const int mb   = wave * 16;            // this wave's 16 batch rows
  const int row  = mb + l15;             // A row (batch) this lane loads
  const int ncol = nb + l15;             // C/D column == W^T row
  const int moff = (lane >> 4) * 8;      // C rows: mb+moff+r

  // One-time cooperative fill of both gates' W^T rows for our 16 columns.
  for (int i = threadIdx.x; i < 16 * NK8; i += 128) {
    const int c  = i / NK8;
    const int k8 = (i - c * NK8) * 8;
    *(v8bf*)&ldsW[0][c][k8] = *(const v8bf*)&WfT[(size_t)(nb + c) * NKTOT + k8];
    *(v8bf*)&ldsW[1][c][k8] = *(const v8bf*)&WhT[(size_t)(nb + c) * NKTOT + k8];
  }
  __syncthreads();

  const bf16_t* arowH = hB + (size_t)row * NDOUT;
  const bf16_t* arowG = gB + (size_t)row * NDOUT;
  const bf16_t* xbase = xt + (size_t)row * NT * NDIN;
  const bf16_t* lbF   = &ldsW[0][l15][0];   // this lane's column, gate f
  const bf16_t* lbH   = &ldsW[1][l15][0];   // this lane's column, gate h

  const float biasF = bfv_[ncol];
  const float biasH = bhv_[ncol];

  const int idxb = (mb + moff) * NDOUT + ncol;          // element this thread owns
  float* outb = out + ((size_t)(mb + moff) * NDOUT + ncol) * NT;

  float hreg[8], freg[8];
  #pragma unroll
  for (int r = 0; r < 8; ++r) hreg[r] = 0.0f;

  unsigned* bar_cnt = bar;
  unsigned* bar_gen = bar + 1;

  // Pre-compute the f-gate x-part for t=0 (barrier-independent).
  v8f accx = gemm_x(xbase, lbF + NDOUT, lane);

  for (int t = 0; t < NT; ++t) {
    const bf16_t* xrow = xbase + (size_t)t * NDIN;
    __builtin_prefetch(xrow + NDIN, 0, 0);              // next step's x row

    // ---- phase 1: f = sigmoid([h, x_t] @ W_f + b_f); g = f*h -> bf16 ----
    v8f acc = gemm_h(arowH, lbF, lane, accx);
    #pragma unroll
    for (int r = 0; r < 8; ++r) {
      const float f = 1.0f / (1.0f + __expf(-(acc[r] + biasF)));
      freg[r] = f;
      gB[idxb + r * NDOUT] = (bf16_t)(f * hreg[r]);
    }
    const unsigned g1 = bar_arrive(bar_cnt, bar_gen);
    accx = gemm_x(xrow, lbH + NDOUT, lane);   // phase-2 x-part hides barrier
    bar_wait(bar_gen, g1);

    // ---- phase 2: h~ = tanh([g, x_t] @ W_h + b_h); h = (1-f)h + f h~ ----
    acc = gemm_h(arowG, lbH, lane, accx);
    #pragma unroll
    for (int r = 0; r < 8; ++r) {
      const float u  = acc[r] + biasH;
      const float e  = __expf(-2.0f * fabsf(u));        // stable tanh
      const float th = copysignf((1.0f - e) / (1.0f + e), u);
      const float hn = (1.0f - freg[r]) * hreg[r] + freg[r] * th;
      hreg[r] = hn;
      hB[idxb + r * NDOUT] = (bf16_t)hn;
      outb[(size_t)r * NDOUT * NT + t] = hn;            // out[b][n][t]
    }
    const unsigned g2 = bar_arrive(bar_cnt, bar_gen);
    const bf16_t* xnext = xbase + (size_t)(t + 1 < NT ? t + 1 : t) * NDIN;
    accx = gemm_x(xnext, lbF + NDOUT, lane);  // next f x-part hides barrier
    bar_wait(bar_gen, g2);
  }
}

// ---------------------------------------------------------------------------
// Prep: transpose+convert W (NKTOT x NDOUT fp32) -> W^T (NDOUT x NKTOT bf16)
// ---------------------------------------------------------------------------
__global__ __launch_bounds__(256) void transpose_w_kernel(
    const float* __restrict__ W, bf16_t* __restrict__ WT)
{
  __shared__ float tile[32][33];
  const int n0 = blockIdx.x * 32;
  const int k0 = blockIdx.y * 32;
  const int tx = threadIdx.x, ty = threadIdx.y;   // (32, 8)
  #pragma unroll
  for (int j = 0; j < 4; ++j)
    tile[ty + 8 * j][tx] = W[(size_t)(k0 + ty + 8 * j) * NDOUT + n0 + tx];
  __syncthreads();
  #pragma unroll
  for (int j = 0; j < 4; ++j)
    WT[(size_t)(n0 + ty + 8 * j) * NKTOT + k0 + tx] = (bf16_t)tile[tx][ty + 8 * j];
}

// Prep: x (B, DIN, T) fp32 -> xt (B, T, DIN) bf16
__global__ __launch_bounds__(256) void transpose_x_kernel(
    const float* __restrict__ x, bf16_t* __restrict__ xt)
{
  __shared__ float tile[32][33];
  const int b  = blockIdx.z;
  const int t0 = blockIdx.x * 32;
  const int d0 = blockIdx.y * 32;
  const int tx = threadIdx.x, ty = threadIdx.y;
  const float* xb = x + (size_t)b * NDIN * NT;
  bf16_t* xtb = xt + (size_t)b * NT * NDIN;
  #pragma unroll
  for (int j = 0; j < 4; ++j)
    tile[ty + 8 * j][tx] = xb[(size_t)(d0 + ty + 8 * j) * NT + t0 + tx];
  __syncthreads();
  #pragma unroll
  for (int j = 0; j < 4; ++j)
    xtb[(size_t)(t0 + ty + 8 * j) * NDIN + d0 + tx] = (bf16_t)tile[tx][ty + 8 * j];
}

__global__ void init_state_kernel(bf16_t* __restrict__ hB, unsigned* __restrict__ bar)
{
  const int i = blockIdx.x * blockDim.x + threadIdx.x;
  if (i < NB * NDOUT) hB[i] = (bf16_t)0.0f;
  if (i < 2) bar[i] = 0u;
}

// ---------------------------------------------------------------------------
extern "C" void kernel_launch(void* const* d_in, const int* in_sizes, int n_in,
                              void* d_out, int out_size, void* d_ws, size_t ws_size,
                              hipStream_t stream)
{
  const float* x  = (const float*)d_in[0];   // (B, DIN, T)
  const float* Wf = (const float*)d_in[1];   // (DIN+DOUT, DOUT)
  const float* bf = (const float*)d_in[2];   // (DOUT,)
  const float* Wh = (const float*)d_in[3];
  const float* bh = (const float*)d_in[4];
  float* out = (float*)d_out;                // (B, DOUT, T)

  char* ws = (char*)d_ws;
  size_t off = 0;
  bf16_t*   WfT = (bf16_t*)(ws + off); off += (size_t)NDOUT * NKTOT * sizeof(bf16_t);
  bf16_t*   WhT = (bf16_t*)(ws + off); off += (size_t)NDOUT * NKTOT * sizeof(bf16_t);
  bf16_t*   xt  = (bf16_t*)(ws + off); off += (size_t)NB * NT * NDIN * sizeof(bf16_t);
  bf16_t*   hB  = (bf16_t*)(ws + off); off += (size_t)NB * NDOUT * sizeof(bf16_t);
  bf16_t*   gB  = (bf16_t*)(ws + off); off += (size_t)NB * NDOUT * sizeof(bf16_t);
  unsigned* bar = (unsigned*)(ws + off); off += 128;
  (void)ws_size; (void)in_sizes; (void)n_in; (void)out_size;

  const dim3 tb(32, 8);
  transpose_w_kernel<<<dim3(NDOUT / 32, NKTOT / 32), tb, 0, stream>>>(Wf, WfT);
  transpose_w_kernel<<<dim3(NDOUT / 32, NKTOT / 32), tb, 0, stream>>>(Wh, WhT);
  transpose_x_kernel<<<dim3(NT / 32, NDIN / 32, NB), tb, 0, stream>>>(x, xt);
  init_state_kernel<<<(NB * NDOUT + 255) / 256, 256, 0, stream>>>(hB, bar);

  // Persistent scan: 64 WGs x 128 threads, all co-resident; 2 split barriers/step.
  mgu_scan_kernel<<<dim3(NWG), 128, 0, stream>>>(
      xt, WfT, WhT, bf, bh, hB, gB, out, bar);
}
